// NetVLAD_78494822301866
// MI455X (gfx1250) — matrix operand order repaired and checked by reading, not compiled, not run on hardware
//
#include <hip/hip_runtime.h>
#include <hip/hip_bf16.h>

#define N_B   32
#define C_DIM 512
#define S_DIM 1024
#define K_CL  64
#define DOUT  256
#define KC    (K_CL * C_DIM)   // 32768

typedef __attribute__((ext_vector_type(16))) _Float16 v16h;
typedef __attribute__((ext_vector_type(8)))  float    v8f;

// ---- WMMA 16x16x32 f16 fragment index maps (CDNA5 ISA 7.12.2, wave32) ----
// A (16x32, MxK): lane<16 -> K in {0..7}U{16..23}; lane>=16 -> +8. VGPR j<4: K=2j(+p); j>=4: K=16+2(j-4)(+p).
__device__ __forceinline__ int a_k_of(int lane, int e) {
  int j = e >> 1, p = e & 1;
  int base = (j < 4) ? (2 * j) : (16 + 2 * (j - 4));
  return base + ((lane & 16) ? 8 : 0) + p;
}
// B (32x16, KxN): same K order as A (lane<16 -> K in {0..7}U{16..23}; lane>=16 -> +8); N = lane&15.
__device__ __forceinline__ int b_k_of(int lane, int e) {
  return a_k_of(lane, e);
}
// D (16x16 f32): VGPR r -> M = r + 8*(lane>=16), N = lane&15.

// ---------------------------------------------------------------- utilities
__global__ void k_zero(float* p, int n) {
  int i = blockIdx.x * 256 + threadIdx.x;
  if (i < n) p[i] = 0.f;
}

__global__ void k_cvt(const float* __restrict__ w, _Float16* __restrict__ o, int n) {
  int i = blockIdx.x * 256 + threadIdx.x;
  if (i < n) o[i] = (_Float16)w[i];
}

// 1/max(||x[n,:,s]||,1e-12), coalesced over s
__global__ void k_rnorm(const float* __restrict__ x, float* __restrict__ rn) {
  int i = blockIdx.x * 256 + threadIdx.x;       // [0, N*S)
  int n = i >> 10, s = i & 1023;
  const float* p = x + (size_t)n * C_DIM * S_DIM + s;
  float acc = 0.f;
  for (int c = 0; c < C_DIM; ++c) { float v = p[(size_t)c * S_DIM]; acc += v * v; }
  rn[i] = 1.f / fmaxf(sqrtf(acc), 1e-12f);
}

// [N][C][S] f32 -> [N][S][C] f16 (scaled), 32x32 LDS tile transpose
__global__ void k_xpose(const float* __restrict__ x, const float* __restrict__ rn,
                        _Float16* __restrict__ xn) {
  int b = blockIdx.x;
  int n = b >> 9; int rem = b & 511;
  int c0 = (rem >> 5) * 32, s0 = (rem & 31) * 32;
  __shared__ float tile[32][33];
  int tid = threadIdx.x;
  #pragma unroll
  for (int r = 0; r < 4; ++r) {
    int idx = tid + r * 256;
    int si = idx & 31, ci = idx >> 5;
    tile[ci][si] = x[((size_t)(n * C_DIM + c0 + ci)) * S_DIM + s0 + si];
  }
  __syncthreads();
  #pragma unroll
  for (int r = 0; r < 4; ++r) {
    int idx = tid + r * 256;
    int ci = idx & 31, si = idx >> 5;
    float v = tile[ci][si] * rn[n * S_DIM + s0 + si];
    xn[((size_t)(n * S_DIM + s0 + si)) * C_DIM + c0 + ci] = (_Float16)v;
  }
}

// ---------------- GEMM1 + softmax:  sa[n,k,s] = softmax_k( conv_w @ xn )
// block = (n, 32-wide s tile); 8 waves = 4 (K-cluster) x 2 (s) 16x16 tiles
#define XPAD 520
__global__ void k_assign(const _Float16* __restrict__ xn,  // [N][S][C]
                         const _Float16* __restrict__ cw,  // [K][C] f16
                         _Float16* __restrict__ sa,        // [N][K][S]
                         float* __restrict__ asum) {       // [N][K]
  const int n = blockIdx.x >> 5;
  const int s0 = (blockIdx.x & 31) * 32;
  const int tid = threadIdx.x, lane = tid & 31, wid = tid >> 5;
  const int mt = wid & 3, st = wid >> 2;

  __shared__ _Float16 xt[32 * XPAD];   // [s_local][c], padded
  __shared__ float    sat[64 * 33];    // [k][s_local], padded

  // stage the whole 32 x 512 xn slab (u32 copies, coalesced)
  {
    const unsigned* g = (const unsigned*)(xn + ((size_t)(n * S_DIM + s0)) * C_DIM);
    for (int t = tid; t < 32 * 256; t += 256) {
      int sl = t >> 8, cu = t & 255;
      ((unsigned*)(xt + sl * XPAD))[cu] = g[sl * 256 + cu];
    }
  }
  __syncthreads();

  v8f acc;
  #pragma unroll
  for (int r = 0; r < 8; ++r) acc[r] = 0.f;
  const int m0 = mt * 16, sl0 = st * 16, am = lane & 15;

  for (int c0 = 0; c0 < C_DIM; c0 += 32) {
    v16h a, bfr;
    #pragma unroll
    for (int e = 0; e < 16; ++e) {
      a[e]   = cw[(m0 + am) * C_DIM + c0 + a_k_of(lane, e)];
      bfr[e] = xt[(sl0 + am) * XPAD + c0 + b_k_of(lane, e)];
    }
    acc = __builtin_amdgcn_wmma_f32_16x16x32_f16(false, a, false, bfr,
                                                 (short)0, acc, false, false);
  }
  #pragma unroll
  for (int r = 0; r < 8; ++r) {
    int m = m0 + r + ((lane & 16) ? 8 : 0);
    sat[m * 33 + sl0 + am] = acc[r];
  }
  __syncthreads();

  // softmax over K=64 for each of 32 s columns
  if (tid < 32) {
    int j = tid;
    float mx = -1e30f;
    for (int k = 0; k < 64; ++k) mx = fmaxf(mx, sat[k * 33 + j]);
    float sum = 0.f;
    for (int k = 0; k < 64; ++k) { float e = expf(sat[k * 33 + j] - mx); sat[k * 33 + j] = e; sum += e; }
    float inv = 1.f / sum;
    for (int k = 0; k < 64; ++k) sat[k * 33 + j] *= inv;
  }
  __syncthreads();
  // coalesced f16 store of probabilities
  for (int t = tid; t < 64 * 32; t += 256) {
    int k = t >> 5, j = t & 31;
    sa[((size_t)(n * K_CL + k)) * S_DIM + s0 + j] = (_Float16)sat[k * 33 + j];
  }
  if (tid < 64) {
    float v = 0.f;
    for (int j = 0; j < 32; ++j) v += sat[tid * 33 + j];
    atomicAdd(&asum[n * K_CL + tid], v);
  }
}

// ---------------- GEMM2: vlad[n,k,c] = sum_s sa*xn - asum*cent
__global__ void k_vlad(const _Float16* __restrict__ sa,   // [N][K][S]
                       const _Float16* __restrict__ xn,   // [N][S][C]
                       const float* __restrict__ asum,    // [N][K]
                       const float* __restrict__ cent,    // [K][C]
                       float* __restrict__ vlad) {        // [N][K][C]
  const int n = blockIdx.x >> 4;
  const int c0 = (blockIdx.x & 15) * 32;
  const int tid = threadIdx.x, lane = tid & 31, wid = tid >> 5;
  const int mt = wid & 3, ct = wid >> 2;

  __shared__ _Float16 satl[64 * 40];  // [k][s_local]
  __shared__ _Float16 xt[32 * 40];    // [s_local][c_local]

  v8f acc;
  #pragma unroll
  for (int r = 0; r < 8; ++r) acc[r] = 0.f;
  const int m0 = mt * 16, cl0 = ct * 16, am = lane & 15;

  for (int s0 = 0; s0 < S_DIM; s0 += 32) {
    __syncthreads();
    for (int t = tid; t < 64 * 32; t += 256) {
      int k = t >> 5, j = t & 31;
      satl[k * 40 + j] = sa[((size_t)(n * K_CL + k)) * S_DIM + s0 + j];
    }
    for (int t = tid; t < 32 * 32; t += 256) {
      int i = t >> 5, j = t & 31;
      xt[i * 40 + j] = xn[((size_t)(n * S_DIM + s0 + i)) * C_DIM + c0 + j];
    }
    __syncthreads();
    v16h a, bfr;
    #pragma unroll
    for (int e = 0; e < 16; ++e) {
      a[e]   = satl[(m0 + am) * 40 + a_k_of(lane, e)];
      bfr[e] = xt[b_k_of(lane, e) * 40 + cl0 + am];
    }
    acc = __builtin_amdgcn_wmma_f32_16x16x32_f16(false, a, false, bfr,
                                                 (short)0, acc, false, false);
  }
  #pragma unroll
  for (int r = 0; r < 8; ++r) {
    int k = m0 + r + ((lane & 16) ? 8 : 0);
    int c = c0 + cl0 + am;
    float v = acc[r] - asum[n * K_CL + k] * cent[k * C_DIM + c];
    vlad[((size_t)(n * K_CL + k)) * C_DIM + c] = v;
  }
}

// ---------------- intra-row norms + global norm accumulation
__global__ void k_norms(const float* __restrict__ vlad,
                        float* __restrict__ rn1, float* __restrict__ gsq) {
  int wid = threadIdx.x >> 5, lane = threadIdx.x & 31;
  int row = blockIdx.x * 8 + wid;   // [0, N*K)
  float s = 0.f;
  #pragma unroll
  for (int t = 0; t < 16; ++t) {
    float v = vlad[(size_t)row * C_DIM + lane + t * 32];
    s += v * v;
  }
  for (int off = 16; off > 0; off >>= 1) s += __shfl_down(s, off, 32);
  if (lane == 0) {
    float r = 1.f / fmaxf(sqrtf(s), 1e-12f);
    rn1[row] = r;
    atomicAdd(&gsq[row >> 6], s * r * r);   // sum of squared intra-normalized rows
  }
}

// ---------------- hidden GEMM: v[32,256] = vladN[32,32768] @ hw[32768,256]
// grid: 16 d-column blocks x 8 K-slabs; 8 waves each reduce 512 Kdim
__global__ void k_hidden(const float* __restrict__ vlad, const float* __restrict__ rn1,
                         const float* __restrict__ gsq, const float* __restrict__ hw,
                         float* __restrict__ vacc) {
  const int d0 = (blockIdx.x & 15) * 16;
  const int slab = blockIdx.x >> 4;               // 0..7
  const int tid = threadIdx.x, lane = tid & 31, wid = tid >> 5;
  const int kbase = slab * 4096 + wid * 512;
  const int am = lane & 15;

  __shared__ _Float16 hwt[8][32][18];   // wave-private B tiles
  __shared__ float    red[8][512];

  v8f acc0, acc1;
  #pragma unroll
  for (int r = 0; r < 8; ++r) { acc0[r] = 0.f; acc1[r] = 0.f; }

  const int row0 = am, row1 = 16 + am;            // batch rows
  const float rg0 = 1.f / fmaxf(sqrtf(gsq[row0]), 1e-12f);
  const float rg1 = 1.f / fmaxf(sqrtf(gsq[row1]), 1e-12f);

  for (int s = 0; s < 16; ++s) {
    const int base = kbase + s * 32;
    if (base + 512 < KC)
      __builtin_prefetch(&hw[(size_t)(base + 512) * DOUT + d0], 0, 0);
    // stage hw tile [32 kdim][16 d], f32 -> f16 on the fly (wave-private)
    for (int q = lane; q < 512; q += 32) {
      int e = q >> 4, j = q & 15;
      hwt[wid][e][j] = (_Float16)hw[(size_t)(base + e) * DOUT + d0 + j];
    }
    asm volatile("s_wait_dscnt 0" ::: "memory");  // wave-private tile: DS in-order per wave

    const int cl = base >> 9;                     // cluster index constant per 32-step
    const float sc0 = rn1[row0 * K_CL + cl] * rg0;
    const float sc1 = rn1[row1 * K_CL + cl] * rg1;
    v16h a0, a1, bfr;
    #pragma unroll
    for (int e = 0; e < 16; ++e) {
      int kk = a_k_of(lane, e);
      a0[e]  = (_Float16)(vlad[(size_t)row0 * KC + base + kk] * sc0);
      a1[e]  = (_Float16)(vlad[(size_t)row1 * KC + base + kk] * sc1);
      bfr[e] = hwt[wid][b_k_of(lane, e)][am];
    }
    acc0 = __builtin_amdgcn_wmma_f32_16x16x32_f16(false, a0, false, bfr,
                                                  (short)0, acc0, false, false);
    acc1 = __builtin_amdgcn_wmma_f32_16x16x32_f16(false, a1, false, bfr,
                                                  (short)0, acc1, false, false);
  }
  #pragma unroll
  for (int r = 0; r < 8; ++r) {
    int m = r + ((lane & 16) ? 8 : 0);
    red[wid][0 * 256 + m * 16 + am] = acc0[r];
    red[wid][1 * 256 + m * 16 + am] = acc1[r];
  }
  __syncthreads();
  for (int q = tid; q < 512; q += 256) {
    float sum = 0.f;
    #pragma unroll
    for (int w = 0; w < 8; ++w) sum += red[w][q];
    int tile = q >> 8, p = q & 255, m = p >> 4, j = p & 15;
    atomicAdd(&vacc[(tile * 16 + m) * DOUT + d0 + j], sum);
  }
}

// ---------------- BN + context gating head (one block, thread = output dim)
__global__ void k_head(const float* __restrict__ vacc, const float* __restrict__ g2,
                       const float* __restrict__ b2, const float* __restrict__ gw,
                       const float* __restrict__ gg, const float* __restrict__ gb,
                       float* __restrict__ out) {
  const int d = threadIdx.x;
  __shared__ float vbn[32][257];
  float col[32];
  float m = 0.f;
  #pragma unroll
  for (int n = 0; n < 32; ++n) { col[n] = vacc[n * DOUT + d]; m += col[n]; }
  m *= (1.f / 32.f);
  float var = 0.f;
  #pragma unroll
  for (int n = 0; n < 32; ++n) { float t = col[n] - m; var += t * t; }
  var *= (1.f / 32.f);
  float rs = rsqrtf(var + 1e-5f);
  float ga = g2[d], be = b2[d];
  #pragma unroll
  for (int n = 0; n < 32; ++n) { col[n] = (col[n] - m) * rs * ga + be; vbn[n][d] = col[n]; }
  __syncthreads();
  float acc[32];
  #pragma unroll
  for (int n = 0; n < 32; ++n) acc[n] = 0.f;
  for (int c = 0; c < 256; ++c) {
    float w = gw[c * DOUT + d];
    #pragma unroll 8
    for (int n = 0; n < 32; ++n) acc[n] += vbn[n][c] * w;
  }
  float m2 = 0.f;
  #pragma unroll
  for (int n = 0; n < 32; ++n) m2 += acc[n];
  m2 *= (1.f / 32.f);
  float v2 = 0.f;
  #pragma unroll
  for (int n = 0; n < 32; ++n) { float t = acc[n] - m2; v2 += t * t; }
  v2 *= (1.f / 32.f);
  float rs2 = rsqrtf(v2 + 1e-5f);
  float gga = gg[d], gbe = gb[d];
  #pragma unroll
  for (int n = 0; n < 32; ++n) {
    float gate = 1.f / (1.f + expf(-((acc[n] - m2) * rs2 * gga + gbe)));
    out[n * DOUT + d] = col[n] * gate;
  }
}

extern "C" void kernel_launch(void* const* d_in, const int* in_sizes, int n_in,
                              void* d_out, int out_size, void* d_ws, size_t ws_size,
                              hipStream_t stream) {
  const float* x    = (const float*)d_in[0];
  const float* cw   = (const float*)d_in[1];
  const float* cent = (const float*)d_in[2];
  const float* hw   = (const float*)d_in[3];
  const float* g2   = (const float*)d_in[4];
  const float* b2   = (const float*)d_in[5];
  const float* gw   = (const float*)d_in[6];
  const float* gg   = (const float*)d_in[7];
  const float* gb   = (const float*)d_in[8];
  float* out = (float*)d_out;

  char* ws = (char*)d_ws;
  size_t off = 0;
  auto alloc = [&](size_t bytes) -> void* {
    void* p = ws + off;
    off += (bytes + 255) & ~(size_t)255;
    return p;
  };
  _Float16* xn16 = (_Float16*)alloc((size_t)N_B * S_DIM * C_DIM * 2);  // 32 MB
  _Float16* cw16 = (_Float16*)alloc((size_t)K_CL * C_DIM * 2);
  _Float16* sa16 = (_Float16*)alloc((size_t)N_B * K_CL * S_DIM * 2);   // 4 MB
  float* vlad = (float*)alloc((size_t)N_B * K_CL * C_DIM * 4);         // 4 MB
  float* rn   = (float*)alloc((size_t)N_B * S_DIM * 4);
  float* zreg = (float*)alloc((size_t)(2048 + 32 + 8192) * 4);         // asum|gsq|vacc
  float* asum = zreg;
  float* gsq  = zreg + 2048;
  float* vacc = zreg + 2048 + 32;
  float* rn1  = (float*)alloc((size_t)2048 * 4);

  k_zero<<<(10272 + 255) / 256, 256, 0, stream>>>(zreg, 10272);
  k_rnorm<<<(N_B * S_DIM) / 256, 256, 0, stream>>>(x, rn);
  k_xpose<<<N_B * 16 * 32, 256, 0, stream>>>(x, rn, xn16);
  k_cvt<<<(K_CL * C_DIM) / 256, 256, 0, stream>>>(cw, cw16, K_CL * C_DIM);
  k_assign<<<N_B * 32, 256, 0, stream>>>(xn16, cw16, sa16, asum);
  k_vlad<<<N_B * 16, 256, 0, stream>>>(sa16, xn16, asum, cent, vlad);
  k_norms<<<(N_B * K_CL) / 8, 256, 0, stream>>>(vlad, rn1, gsq);
  k_hidden<<<16 * 8, 256, 0, stream>>>(vlad, rn1, gsq, hw, vacc);
  k_head<<<1, 256, 0, stream>>>(vacc, g2, b2, gw, gg, gb, out);
}
